// AttentionLayer_77017353552431
// MI455X (gfx1250) — matrix operand order repaired and checked by reading, not compiled, not run on hardware
//
#include <hip/hip_runtime.h>
#include <hip/hip_bf16.h>

// ---------------------------------------------------------------------------
// Problem constants (reference: B=2, S=4096, D=512, H=8, E=64, all f32 I/O)
// ---------------------------------------------------------------------------
#define BB 2
#define SS 4096
#define DD 512
#define HH 8
#define EE 64

typedef __attribute__((ext_vector_type(16))) _Float16     v16h;
typedef __attribute__((ext_vector_type(8)))  _Float16     v8h;
typedef __attribute__((ext_vector_type(8)))  float        v8f;
typedef __attribute__((ext_vector_type(4)))  unsigned int u32x4;
typedef __attribute__((ext_vector_type(8)))  unsigned int u32x8;

static __device__ __forceinline__ v8f zero8() {
  v8f z = {0.f,0.f,0.f,0.f,0.f,0.f,0.f,0.f};
  return z;
}

// ---------------------------------------------------------------------------
// WMMA fragment loaders (CDNA5 16x16x32 f16, wave32 layouts per ISA 7.12.2)
// A (16x32, row-major source, ld in halves):
//   lanes 0-15 : row = lane,    halves[0..7]=K0..7,  halves[8..15]=K16..23
//   lanes 16-31: row = lane-16, halves[0..7]=K8..15, halves[8..15]=K24..31
// ---------------------------------------------------------------------------
static __device__ __forceinline__ v16h frag_a_rm(const _Float16* tile, int ld) {
  const int lane = threadIdx.x & 31;
  const _Float16* p = tile + (lane & 15) * ld + ((lane >> 4) << 3);
  v8h lo = *reinterpret_cast<const v8h*>(p);
  v8h hi = *reinterpret_cast<const v8h*>(p + 16);
  v16h a;
#pragma unroll
  for (int i = 0; i < 8; ++i) { a[i] = lo[i]; a[i + 8] = hi[i]; }
  return a;
}

// B (32x16) from packed column-major storage Bp[N][K] (ld = K in halves):
//   lanes 0-15 : col N = lane,    halves[0..7]=K0..7,  halves[8..15]=K16..23
//   lanes 16-31: col N = lane-16, halves[0..7]=K8..15, halves[8..15]=K24..31
static __device__ __forceinline__ v16h frag_b_cm(const _Float16* tile, int ld) {
  const int lane = threadIdx.x & 31;
  const _Float16* p = tile + (size_t)(lane & 15) * ld + ((lane >> 4) << 3);
  v8h lo = *reinterpret_cast<const v8h*>(p);
  v8h hi = *reinterpret_cast<const v8h*>(p + 16);
  v16h b;
#pragma unroll
  for (int i = 0; i < 8; ++i) { b[i] = lo[i]; b[i + 8] = hi[i]; }
  return b;
}

#define WMMA_F16(Afr, Bfr, Cfr) \
  __builtin_amdgcn_wmma_f32_16x16x32_f16(false, (Afr), false, (Bfr), (short)0, (Cfr), false, false)

// ---------------------------------------------------------------------------
// Tensor Data Mover descriptor packing (CDNA5 ISA)
// Group 0: [1:0]count=1 | [63:32]lds_addr | [120:64]global_addr | [127:126]=2
// ---------------------------------------------------------------------------
static __device__ __forceinline__ u32x4 tdm_group0(unsigned lds_addr,
                                                   unsigned long long gaddr) {
  u32x4 g;
  g[0] = 1u;                                       // count=1, user descriptor
  g[1] = lds_addr;                                 // LDS byte address
  g[2] = (unsigned)(gaddr & 0xFFFFFFFFu);          // global_addr[31:0]
  g[3] = (unsigned)((gaddr >> 32) & 0x1FFFFFFu)    // global_addr[56:32]
       | (2u << 30);                               // type = 2 ("image")
  return g;
}

// Group 1 for a 2D tile of 16-bit elements (data_size code 1):
//  [17:16]=1 | [79:48]tensor_dim0 | [111:80]tensor_dim1 | [127:112]tile_dim0
//  [143:128]tile_dim1 | [207:160]tensor_dim0_stride  (all element units)
static __device__ __forceinline__ u32x8 tdm_group1_2d(unsigned tensor_d0,
                                                      unsigned tensor_d1,
                                                      unsigned tile_d0,
                                                      unsigned tile_d1,
                                                      unsigned long long d0_stride) {
  u32x8 g;
  g[0] = 1u << 16;                                           // data_size = 2B
  g[1] = (tensor_d0 & 0xFFFFu) << 16;                        // dim0[15:0]
  g[2] = (tensor_d0 >> 16) | ((tensor_d1 & 0xFFFFu) << 16);  // dim0[31:16]|dim1[15:0]
  g[3] = (tensor_d1 >> 16) | ((tile_d0 & 0xFFFFu) << 16);    // dim1[31:16]|tile0
  g[4] = tile_d1 & 0xFFFFu;                                  // tile1 (tile2=0)
  g[5] = (unsigned)(d0_stride & 0xFFFFFFFFu);                // stride0[31:0]
  g[6] = (unsigned)((d0_stride >> 32) & 0xFFFFu);            // stride0[47:32]
  g[7] = 0u;                                                 // dim1_stride unused
  return g;
}

static __device__ __forceinline__ void tdm_load(const u32x4& g0, const u32x8& g1) {
  asm volatile("tensor_load_to_lds %0, %1" :: "s"(g0), "s"(g1) : "memory");
}

// ---------------------------------------------------------------------------
// Generic WMMA GEMM:  C[M,N] = act(A[M,K](f16) * Bp[N,K]^T(f16) + bias)
// 256 threads = 8 waves -> 32(M) x 64(N) per block.  A is staged into LDS by
// GLOBAL_LOAD_ASYNC_TO_LDS_B128 (ASYNCcnt), double-buffered so the async copy
// of chunk k+1 overlaps the WMMAs of chunk k.  B fragments stream from packed
// global (L2-resident), with a near-cache prefetch one chunk ahead.
// mode: 0 = f16 row-major out, 1 = f16 transposed out[n*ldc+m], 2 = f32 out.
// ---------------------------------------------------------------------------
__global__ __launch_bounds__(256) void gemm_wmma(
    const _Float16* __restrict__ A, const _Float16* __restrict__ Bp,
    const float* __restrict__ bias, void* __restrict__ out,
    int K, int ldc, int mode, int relu) {
  __shared__ __attribute__((aligned(16))) _Float16 As[2][32 * 72];

  const int m0   = blockIdx.x * 32;
  const int n0   = blockIdx.y * 64;
  const int tid  = threadIdx.x;
  const int wave = tid >> 5;
  const int lane = tid & 31;
  const int mt   = wave >> 2;   // 0..1
  const int nt   = wave & 3;    // 0..3
  const int ntn  = n0 + nt * 16;

  v8f c = zero8();

  const int lrow = tid >> 3;        // 0..31  (one b128 per thread per chunk)
  const int lcol = (tid & 7) * 8;   // 0..56

  // async copy of one 32x64 A chunk straight into LDS (no VGPR round-trip)
  auto issue_a = [&](int buf, int kb) {
    const unsigned lds_off =
        (unsigned)(size_t)&As[buf][lrow * 72 + lcol];
    const unsigned long long ga =
        (unsigned long long)(size_t)(A + (size_t)(m0 + lrow) * K + kb + lcol);
    asm volatile("global_load_async_to_lds_b128 %0, %1, off"
                 :: "v"(lds_off), "v"(ga) : "memory");
  };

  issue_a(0, 0);
  int cur = 0;
  for (int k0 = 0; k0 < K; k0 += 64) {
    if (k0 + 64 < K) {
      issue_a(cur ^ 1, k0 + 64);                         // overlap next copy
      __builtin_prefetch(Bp + (size_t)ntn * K + k0 + 64, 0, 3);
      asm volatile("s_wait_asynccnt 0x1" ::: "memory");  // oldest chunk landed
    } else {
      asm volatile("s_wait_asynccnt 0x0" ::: "memory");
    }
    __syncthreads();
#pragma unroll
    for (int kk = 0; kk < 64; kk += 32) {
      v16h aF = frag_a_rm(&As[cur][(mt * 16) * 72 + kk], 72);
      v16h bF = frag_b_cm(Bp + (size_t)ntn * K + k0 + kk, K);
      c = WMMA_F16(aF, bF, c);
    }
    __syncthreads();   // everyone done with As[cur] before it is re-filled
    cur ^= 1;
  }

  // Epilogue: C element (m = r + 8*(lane>=16) + tile, n = lane&15 + tile)
  const int nn    = ntn + (lane & 15);
  const int mbase = m0 + mt * 16 + ((lane >> 4) << 3);
  const float bv  = bias ? bias[nn] : 0.0f;
#pragma unroll
  for (int r = 0; r < 8; ++r) {
    float v = c[r] + bv;
    if (relu) v = fmaxf(v, 0.0f);
    const int mm = mbase + r;
    if (mode == 0)      ((_Float16*)out)[(size_t)mm * ldc + nn] = (_Float16)v;
    else if (mode == 1) ((_Float16*)out)[(size_t)nn * ldc + mm] = (_Float16)v;
    else                ((float*)out)[(size_t)mm * ldc + nn] = v;
  }
}

// ---------------------------------------------------------------------------
// Flash attention, one (b,h) per blockIdx.y, 4 waves x 16 query rows / block.
// All 4 waves share the same 32-key K/V tiles, so wave 0 stages them with the
// Tensor Data Mover (tensor_load_to_lds, TENSORcnt), double-buffered:
//   K tile : 32 rows x 64 el  from K[bh][S][E]   (contiguous rows)
//   V tile : 64 rows x 32 el  from V^T[bh][E][S] (rows strided by S -> 2D DMA)
// Scores and P*V are 8 WMMAs per key block with online softmax; P round-trips
// through a per-wave LDS tile to become an A fragment.
// ---------------------------------------------------------------------------
__global__ __launch_bounds__(128) void flash_attn(
    const _Float16* __restrict__ Qh, const _Float16* __restrict__ Kh,
    const _Float16* __restrict__ VT, _Float16* __restrict__ Ctx) {
  __shared__ __attribute__((aligned(16))) _Float16 Klds[2][32 * 64];
  __shared__ __attribute__((aligned(16))) _Float16 Vlds[2][64 * 32];
  __shared__ __attribute__((aligned(16))) _Float16 Plds[4][16 * 32];

  const int wave = threadIdx.x >> 5;
  const int lane = threadIdx.x & 31;
  const int bh   = blockIdx.y;
  const int b    = bh >> 3;
  const int h    = bh & 7;
  const int q0   = blockIdx.x * 64 + wave * 16;

  const _Float16* Qb = Qh + (size_t)bh * SS * EE;
  const _Float16* Kb = Kh + (size_t)bh * SS * EE;
  const _Float16* Vb = VT + (size_t)bh * EE * SS;
  _Float16* Pw = &Plds[wave][0];

  // TDM tile descriptors (group1 is loop-invariant)
  const u32x8 gK1 = tdm_group1_2d(/*tensor_d0=*/EE, /*tensor_d1=*/SS,
                                  /*tile_d0=*/EE, /*tile_d1=*/32,
                                  /*d0_stride=*/EE);
  const u32x8 gV1 = tdm_group1_2d(/*tensor_d0=*/SS, /*tensor_d1=*/EE,
                                  /*tile_d0=*/32, /*tile_d1=*/EE,
                                  /*d0_stride=*/SS);

  auto issue_kv = [&](int buf, int t0) {
    u32x4 gK0 = tdm_group0((unsigned)(size_t)&Klds[buf][0],
                           (unsigned long long)(size_t)(Kb + (size_t)t0 * EE));
    u32x4 gV0 = tdm_group0((unsigned)(size_t)&Vlds[buf][0],
                           (unsigned long long)(size_t)(Vb + t0));
    tdm_load(gK0, gK1);
    tdm_load(gV0, gV1);
  };

  // Q fragments live in registers for the whole key sweep (K-dim = E = 64)
  const v16h aQ0 = frag_a_rm(Qb + (size_t)q0 * EE + 0, EE);
  const v16h aQ1 = frag_a_rm(Qb + (size_t)q0 * EE + 32, EE);

  v8f acc0 = zero8(), acc1 = zero8(), acc2 = zero8(), acc3 = zero8();
  float mrow[8], lrow[8];
#pragma unroll
  for (int r = 0; r < 8; ++r) { mrow[r] = -1e30f; lrow[r] = 0.f; }

  const float scale = 0.125f;  // 1/sqrt(64)

  if (wave == 0) issue_kv(0, 0);
  int cur = 0;

  for (int t0 = 0; t0 < SS; t0 += 32) {
    if (wave == 0) {
      if (t0 + 32 < SS) {
        issue_kv(cur ^ 1, t0 + 32);                         // overlap next DMA
        asm volatile("s_wait_tensorcnt 0x2" ::: "memory");  // current pair done
      } else {
        asm volatile("s_wait_tensorcnt 0x0" ::: "memory");
      }
    }
    __syncthreads();
    const _Float16* Kt = &Klds[cur][0];
    const _Float16* Vt = &Vlds[cur][0];

    // scores: two 16x16 tiles over this 32-key block, K-dim = 64
    v8f s0 = zero8(), s1 = zero8();
    s0 = WMMA_F16(aQ0, frag_b_cm(Kt + 0 * 64 + 0, 64), s0);
    s0 = WMMA_F16(aQ1, frag_b_cm(Kt + 0 * 64 + 32, 64), s0);
    s1 = WMMA_F16(aQ0, frag_b_cm(Kt + 16 * 64 + 0, 64), s1);
    s1 = WMMA_F16(aQ1, frag_b_cm(Kt + 16 * 64 + 32, 64), s1);

    float p0[8], p1[8];
#pragma unroll
    for (int r = 0; r < 8; ++r) {
      float x0 = s0[r] * scale, x1 = s1[r] * scale;
      float mx = fmaxf(x0, x1);                 // row max across 16 lanes
      mx = fmaxf(mx, __shfl_xor(mx, 1));
      mx = fmaxf(mx, __shfl_xor(mx, 2));
      mx = fmaxf(mx, __shfl_xor(mx, 4));
      mx = fmaxf(mx, __shfl_xor(mx, 8));
      const float mnew  = fmaxf(mrow[r], mx);
      const float alpha = __expf(mrow[r] - mnew);
      mrow[r] = mnew;
      const float e0 = __expf(x0 - mnew);
      const float e1 = __expf(x1 - mnew);
      p0[r] = e0; p1[r] = e1;
      float rs = e0 + e1;                       // row sum across 16 lanes
      rs += __shfl_xor(rs, 1); rs += __shfl_xor(rs, 2);
      rs += __shfl_xor(rs, 4); rs += __shfl_xor(rs, 8);
      lrow[r] = lrow[r] * alpha + rs;
      acc0[r] *= alpha; acc1[r] *= alpha; acc2[r] *= alpha; acc3[r] *= alpha;
    }

    // P (C layout) -> per-wave LDS 16x32 so it can be re-read as an A fragment
    {
      const int col   = lane & 15;
      const int rbase = (lane >> 4) << 3;
#pragma unroll
      for (int r = 0; r < 8; ++r) {
        Pw[(rbase + r) * 32 + col]      = (_Float16)p0[r];
        Pw[(rbase + r) * 32 + 16 + col] = (_Float16)p1[r];
      }
    }
    asm volatile("s_wait_dscnt 0x0" ::: "memory");  // DS is in-order per wave

    const v16h aP = frag_a_rm(Pw, 32);
    acc0 = WMMA_F16(aP, frag_b_cm(Vt + (0 * 16) * 32, 32), acc0);
    acc1 = WMMA_F16(aP, frag_b_cm(Vt + (1 * 16) * 32, 32), acc1);
    acc2 = WMMA_F16(aP, frag_b_cm(Vt + (2 * 16) * 32, 32), acc2);
    acc3 = WMMA_F16(aP, frag_b_cm(Vt + (3 * 16) * 32, 32), acc3);

    __syncthreads();   // everyone done with K/V[cur] before TDM refills it
    cur ^= 1;
  }

  // normalize and emit ctx[b][s][h*64+e]
  const int col   = lane & 15;
  const int mbase = q0 + ((lane >> 4) << 3);
#pragma unroll
  for (int r = 0; r < 8; ++r) {
    const float inv = 1.0f / lrow[r];
    const size_t ro = ((size_t)(b * SS + mbase + r)) * (HH * EE) + h * EE;
    Ctx[ro + 0 * 16 + col] = (_Float16)(acc0[r] * inv);
    Ctx[ro + 1 * 16 + col] = (_Float16)(acc1[r] * inv);
    Ctx[ro + 2 * 16 + col] = (_Float16)(acc2[r] * inv);
    Ctx[ro + 3 * 16 + col] = (_Float16)(acc3[r] * inv);
  }
}

// ---------------------------------------------------------------------------
// Pack / convert helpers
// ---------------------------------------------------------------------------
__global__ void k_cvt_f16(const float* __restrict__ in, _Float16* __restrict__ out, int n) {
  for (int i = blockIdx.x * blockDim.x + threadIdx.x; i < n; i += gridDim.x * blockDim.x)
    out[i] = (_Float16)in[i];
}

// in: [batch][rows][cols] f32  ->  out: [batch][cols][rows] f16
__global__ void k_pack_T(const float* __restrict__ in, _Float16* __restrict__ out,
                         int rows, int cols) {
  const float* ib = in + (size_t)blockIdx.y * rows * cols;
  _Float16*   ob = out + (size_t)blockIdx.y * rows * cols;
  const int total = rows * cols;
  for (int i = blockIdx.x * blockDim.x + threadIdx.x; i < total; i += gridDim.x * blockDim.x) {
    const int r = i / cols, cI = i % cols;
    ob[(size_t)cI * rows + r] = (_Float16)ib[i];
  }
}

// wo: [H][E][D] f32 -> WOT: [D][H*E] f16  (column-major pack of stacked wo)
__global__ void k_pack_wo(const float* __restrict__ wo, _Float16* __restrict__ out) {
  const int total = HH * EE * DD;
  for (int i = blockIdx.x * blockDim.x + threadIdx.x; i < total; i += gridDim.x * blockDim.x) {
    const int hI  = i / (EE * DD);
    const int rem = i % (EE * DD);
    const int e = rem / DD, d = rem % DD;
    out[(size_t)d * (HH * EE) + hI * EE + e] = (_Float16)wo[i];
  }
}

__global__ void k_bo_sum(const float* __restrict__ bo, float* __restrict__ out) {
  const int d = blockIdx.x * blockDim.x + threadIdx.x;
  if (d < DD) {
    float s = 0.f;
    for (int hI = 0; hI < HH; ++hI) s += bo[hI * DD + d];
    out[d] = s;
  }
}

// ---------------------------------------------------------------------------
// Host launcher
// ---------------------------------------------------------------------------
extern "C" void kernel_launch(void* const* d_in, const int* in_sizes, int n_in,
                              void* d_out, int out_size, void* d_ws, size_t ws_size,
                              hipStream_t stream) {
  (void)in_sizes; (void)n_in; (void)out_size; (void)ws_size;
  const float* x  = (const float*)d_in[0];
  const float* wq = (const float*)d_in[1];
  const float* bq = (const float*)d_in[2];
  const float* wk = (const float*)d_in[3];
  const float* bk = (const float*)d_in[4];
  const float* wv = (const float*)d_in[5];
  const float* bv = (const float*)d_in[6];
  const float* wo = (const float*)d_in[7];
  const float* bo = (const float*)d_in[8];
  const float* w1 = (const float*)d_in[9];
  const float* b1 = (const float*)d_in[10];
  const float* w2 = (const float*)d_in[11];
  const float* b2 = (const float*)d_in[12];

  char* ws = (char*)d_ws;
  size_t off = 0;
  auto alloc = [&](size_t bytes) -> void* {
    void* p = ws + off;
    off += (bytes + 255) & ~(size_t)255;
    return p;
  };

  _Float16* xh    = (_Float16*)alloc((size_t)BB * SS * DD * 2);
  _Float16* wqT   = (_Float16*)alloc((size_t)HH * EE * DD * 2);
  _Float16* wkT   = (_Float16*)alloc((size_t)HH * EE * DD * 2);
  _Float16* wvT   = (_Float16*)alloc((size_t)HH * EE * DD * 2);
  _Float16* WOT   = (_Float16*)alloc((size_t)DD * HH * EE * 2);
  _Float16* w1T   = (_Float16*)alloc((size_t)DD * DD * 2);
  _Float16* w2T   = (_Float16*)alloc((size_t)DD * DD * 2);
  float*    boS   = (float*)   alloc((size_t)DD * 4);
  _Float16* qbuf  = (_Float16*)alloc((size_t)BB * HH * SS * EE * 2);
  _Float16* kbuf  = (_Float16*)alloc((size_t)BB * HH * SS * EE * 2);
  _Float16* vtbuf = (_Float16*)alloc((size_t)BB * HH * EE * SS * 2);  // V^T
  _Float16* ctx   = (_Float16*)alloc((size_t)BB * SS * HH * EE * 2);
  _Float16* ybuf  = (_Float16*)alloc((size_t)BB * SS * DD * 2);
  _Float16* hbuf  = (_Float16*)alloc((size_t)BB * SS * DD * 2);

  // --- pack / convert ---
  k_cvt_f16<<<4096, 256, 0, stream>>>(x, xh, BB * SS * DD);
  k_pack_T<<<dim3(128, HH), 256, 0, stream>>>(wq, wqT, DD, EE);
  k_pack_T<<<dim3(128, HH), 256, 0, stream>>>(wk, wkT, DD, EE);
  k_pack_T<<<dim3(128, HH), 256, 0, stream>>>(wv, wvT, DD, EE);
  k_pack_wo<<<512, 256, 0, stream>>>(wo, WOT);
  k_pack_T<<<dim3(512, 1), 256, 0, stream>>>(w1, w1T, DD, DD);
  k_pack_T<<<dim3(512, 1), 256, 0, stream>>>(w2, w2T, DD, DD);
  k_bo_sum<<<2, 256, 0, stream>>>(bo, boS);

  // --- QKV projections: per (b,h) 4096x64 = 4096x512 @ 512x64 ---
  for (int b = 0; b < BB; ++b) {
    for (int h = 0; h < HH; ++h) {
      const int bh = b * HH + h;
      const _Float16* Ab = xh + (size_t)b * SS * DD;
      gemm_wmma<<<dim3(SS / 32, 1), 256, 0, stream>>>(
          Ab, wqT + (size_t)h * EE * DD, bq + h * EE,
          qbuf + (size_t)bh * SS * EE, DD, EE, /*mode=*/0, 0);
      gemm_wmma<<<dim3(SS / 32, 1), 256, 0, stream>>>(
          Ab, wkT + (size_t)h * EE * DD, bk + h * EE,
          kbuf + (size_t)bh * SS * EE, DD, EE, /*mode=*/0, 0);
      gemm_wmma<<<dim3(SS / 32, 1), 256, 0, stream>>>(
          Ab, wvT + (size_t)h * EE * DD, bv + h * EE,
          vtbuf + (size_t)bh * EE * SS, DD, /*ldc=*/SS, /*mode=*/1, 0);  // V^T
    }
  }

  // --- flash attention ---
  flash_attn<<<dim3(SS / 64, BB * HH), 128, 0, stream>>>(qbuf, kbuf, vtbuf, ctx);

  // --- output projection: [B*S,512] @ [512,512] + sum(bo) ---
  gemm_wmma<<<dim3(BB * SS / 32, DD / 64), 256, 0, stream>>>(
      ctx, WOT, boS, ybuf, HH * EE, DD, /*mode=*/0, 0);

  // --- MLP ---
  gemm_wmma<<<dim3(BB * SS / 32, DD / 64), 256, 0, stream>>>(
      ybuf, w1T, b1, hbuf, DD, DD, /*mode=*/0, /*relu=*/1);
  gemm_wmma<<<dim3(BB * SS / 32, DD / 64), 256, 0, stream>>>(
      hbuf, w2T, b2, d_out, DD, DD, /*mode=*/2, 0);
}
